// DecoderBlock_2482491097850
// MI455X (gfx1250) — compile-verified, matrix operand order checked by reading, not run on hardware
//
#include <hip/hip_runtime.h>
#include <hip/hip_bf16.h>
#include <math.h>
#include <stdint.h>

// ---------------- problem constants ----------------
#define B_    8
#define N_TOK 1024
#define C_    768
#define H_    12
#define D_    64
#define HID_  3072

typedef __bf16 bf16_t;
typedef bf16_t bf16x16 __attribute__((ext_vector_type(16)));
typedef float  f32x8   __attribute__((ext_vector_type(8)));
typedef unsigned int u32x4 __attribute__((ext_vector_type(4)));
typedef int          i32x4 __attribute__((ext_vector_type(4)));
typedef int          i32x8 __attribute__((ext_vector_type(8)));

#if defined(__has_builtin)
#if __has_builtin(__builtin_amdgcn_tensor_load_to_lds) && \
    __has_builtin(__builtin_amdgcn_s_wait_tensorcnt)
#define USE_TDM 1
#endif
#endif

union FragBF {
  bf16x16        v;
  unsigned int   u[8];
  unsigned short s[16];
};
union FragF {
  f32x8 v;
  float f[8];
};

__device__ __forceinline__ unsigned short f2bf(float f) {
  union { float f; unsigned int u; } a; a.f = f;
  unsigned int r = a.u + 0x7FFFu + ((a.u >> 16) & 1u);   // round-to-nearest-even
  return (unsigned short)(r >> 16);
}

// CDNA5 16-bit A/B fragment layout (16x32 / 32x16):
//   lane = half*16 + m ; element e holds K = kb + (e>>3)*16 + half*8 + (e&7)
// For row-major [row][K] storage this is two contiguous 16-byte chunks per lane.
__device__ __forceinline__ FragBF load_frag(const unsigned short* base, int ld,
                                            int row, int kb, int half) {
  const uint4* p = reinterpret_cast<const uint4*>(base + (long)row * ld + kb + half * 8);
  uint4 c0 = p[0];
  uint4 c1 = p[2];   // +32 bytes = +16 elements
  FragBF f;
  f.u[0] = c0.x; f.u[1] = c0.y; f.u[2] = c0.z; f.u[3] = c0.w;
  f.u[4] = c1.x; f.u[5] = c1.y; f.u[6] = c1.z; f.u[7] = c1.w;
  return f;
}

// Same fragment gather from an LDS row (row stride must keep 16B alignment).
__device__ __forceinline__ FragBF load_frag_lds_row(const unsigned short* rowp, int half) {
  const uint4* p = reinterpret_cast<const uint4*>(rowp + half * 8);
  uint4 c0 = p[0];
  uint4 c1 = p[2];   // +16 elements
  FragBF f;
  f.u[0] = c0.x; f.u[1] = c0.y; f.u[2] = c0.z; f.u[3] = c0.w;
  f.u[4] = c1.x; f.u[5] = c1.y; f.u[6] = c1.z; f.u[7] = c1.w;
  return f;
}

__device__ __forceinline__ f32x8 wmma_bf16(const FragBF& a, const FragBF& b, f32x8 c) {
  return __builtin_amdgcn_wmma_f32_16x16x32_bf16(false, a.v, false, b.v,
                                                 (short)0, c, false, false);
}

// ---------------- weight convert + transpose: Wt[n][k] = bf16(W[k][n]) ----------
__global__ __launch_bounds__(256)
void cvtT_kernel(const float* __restrict__ W, unsigned short* __restrict__ Wt,
                 int K, int N) {
  long idx = (long)blockIdx.x * 256 + threadIdx.x;   // over N*K, k fastest
  int k = (int)(idx % K);
  int n = (int)(idx / K);
  Wt[idx] = f2bf(W[(long)k * N + n]);
}

// ---------------- LayerNorm (row = one block of 256, C=768) ---------------------
__global__ __launch_bounds__(256)
void ln_kernel(const float* __restrict__ x, const float* __restrict__ g,
               const float* __restrict__ bta, unsigned short* __restrict__ out) {
  const int row = blockIdx.x;
  const int tid = threadIdx.x;
  const float* xr = x + (long)row * C_;
  float v0 = xr[tid], v1 = xr[tid + 256], v2 = xr[tid + 512];
  float s = v0 + v1 + v2;
  float q = v0 * v0 + v1 * v1 + v2 * v2;
  #pragma unroll
  for (int m = 1; m < 32; m <<= 1) {
    s += __shfl_xor(s, m, 32);
    q += __shfl_xor(q, m, 32);
  }
  __shared__ float ss[8], sq[8];
  if ((tid & 31) == 0) { ss[tid >> 5] = s; sq[tid >> 5] = q; }
  __syncthreads();
  float ts = 0.f, tq = 0.f;
  #pragma unroll
  for (int i = 0; i < 8; i++) { ts += ss[i]; tq += sq[i]; }
  float mu  = ts * (1.0f / C_);
  float var = tq * (1.0f / C_) - mu * mu;
  float rs  = rsqrtf(var + 1e-5f);
  unsigned short* orow = out + (long)row * C_;
  orow[tid]       = f2bf((v0 - mu) * rs * g[tid]       + bta[tid]);
  orow[tid + 256] = f2bf((v1 - mu) * rs * g[tid + 256] + bta[tid + 256]);
  orow[tid + 512] = f2bf((v2 - mu) * rs * g[tid + 512] + bta[tid + 512]);
}

// ---------------- bf16 GEMM: out = A(MxK) * Bt(NxK)^T, templated epilogue -------
// EPI 0: QKV split -> q/k/v (B,H,N,D) bf16
// EPI 1: outF = resid + (acc + bias) * ls      (fp32)
// EPI 2: outQ = bf16(gelu_exact(acc + bias))
template <int EPI>
__global__ __launch_bounds__(256)
void gemm_kernel(const unsigned short* __restrict__ A,
                 const unsigned short* __restrict__ Bt,
                 int M, int N, int K,
                 const float* __restrict__ bias, const float* __restrict__ ls,
                 const float* __restrict__ resid, float* __restrict__ outF,
                 unsigned short* __restrict__ outQ, unsigned short* __restrict__ outK,
                 unsigned short* __restrict__ outV) {
  const int tid  = threadIdx.x;
  const int w    = tid >> 5;
  const int lane = tid & 31;
  const int half = lane >> 4;
  const int lm   = lane & 15;
  const int wm   = w & 1, wn = w >> 1;              // 2 x 4 wave grid
  const int gm   = blockIdx.y * 64 + wm * 32;
  const int gn   = blockIdx.x * 128 + wn * 32;

  FragF acc[2][2];
  #pragma unroll
  for (int i = 0; i < 2; i++)
    #pragma unroll
    for (int j = 0; j < 2; j++)
      #pragma unroll
      for (int e = 0; e < 8; e++) acc[i][j].f[e] = 0.0f;

  for (int kb = 0; kb < K; kb += 32) {
    // prefetch next K-slab (emits global_prefetch_b8; cheap, L2 is 192MB)
    if (kb + 32 < K) {
      __builtin_prefetch(A  + (long)(gm + lm) * K + kb + 32, 0, 0);
      __builtin_prefetch(Bt + (long)(gn + lm) * K + kb + 32, 0, 0);
    }
    FragBF a0 = load_frag(A,  K, gm + lm,      kb, half);
    FragBF a1 = load_frag(A,  K, gm + 16 + lm, kb, half);
    FragBF b0 = load_frag(Bt, K, gn + lm,      kb, half);
    FragBF b1 = load_frag(Bt, K, gn + 16 + lm, kb, half);
    acc[0][0].v = wmma_bf16(a0, b0, acc[0][0].v);
    acc[0][1].v = wmma_bf16(a0, b1, acc[0][1].v);
    acc[1][0].v = wmma_bf16(a1, b0, acc[1][0].v);
    acc[1][1].v = wmma_bf16(a1, b1, acc[1][1].v);
  }

  #pragma unroll
  for (int i = 0; i < 2; i++) {
    #pragma unroll
    for (int j = 0; j < 2; j++) {
      #pragma unroll
      for (int v = 0; v < 8; v++) {
        int row = gm + i * 16 + v + 8 * half;   // C/D layout: row = vgpr + 8*half
        int col = gn + j * 16 + lm;             //             col = lane & 15
        float val = acc[i][j].f[v];
        if (EPI == 0) {
          int which = col / C_;
          int cc = col - which * C_;
          int hh = cc >> 6;
          int d  = cc & 63;
          int bb = row >> 10;
          int nn = row & 1023;
          long idx = (((long)bb * H_ + hh) * N_TOK + nn) * D_ + d;
          unsigned short bv = f2bf(val);
          if (which == 0)      outQ[idx] = bv;
          else if (which == 1) outK[idx] = bv;
          else                 outV[idx] = bv;
        } else if (EPI == 1) {
          long idx = (long)row * N + col;
          outF[idx] = resid[idx] + (val + bias[col]) * ls[col];
        } else {
          float t = val + bias[col];
          float gl = 0.5f * t * (1.0f + erff(t * 0.70710678118654752f));
          outQ[(long)row * N + col] = f2bf(gl);
        }
      }
    }
  }
}

// ---------------- flash attention: one (b, h, 128-query block) per workgroup ----
__global__ __launch_bounds__(256)
void attn_kernel(const unsigned short* __restrict__ Q,
                 const unsigned short* __restrict__ Kt,
                 const unsigned short* __restrict__ V,
                 const float* __restrict__ temperature,
                 const float* __restrict__ locw,
                 unsigned short* __restrict__ out) {
#ifdef USE_TDM
  __shared__ __align__(16) unsigned short Vraw[32][64];   // V tile [key][d] via TDM
#else
  __shared__ __align__(16) unsigned short Vt[64][40];     // V tile transposed [d][key]
#endif
  __shared__ __align__(16) unsigned short Pbuf[8][16][40];  // per-wave P staging [row][key]

  const int tid  = threadIdx.x;
  const int w    = tid >> 5;
  const int lane = tid & 31;
  const int half = lane >> 4;
  const int lm   = lane & 15;

  const int bx   = blockIdx.x;
  const int qblk = bx & 7;                 // N/128 = 8
  const int h    = (bx >> 3) % H_;
  const int b    = bx / (8 * H_);

  const long headoff = ((long)b * H_ + h) * (long)N_TOK * D_;
  const unsigned short* Qh = Q  + headoff;
  const unsigned short* Kh = Kt + headoff;
  const unsigned short* Vh = V  + headoff;

  const int   qbase = qblk * 128 + w * 16;
  const float scale = __expf(temperature[h]);
  const float lw    = locw[h];
  const float inv31 = 1.0f / 31.0f;

  FragBF qa[2];
  qa[0] = load_frag(Qh, D_, qbase + lm, 0,  half);
  qa[1] = load_frag(Qh, D_, qbase + lm, 32, half);

  float yi[8], xi[8];
  #pragma unroll
  for (int v = 0; v < 8; v++) {
    int n = qbase + v + 8 * half;
    yi[v] = (float)(n >> 5) * inv31;
    xi[v] = (float)(n & 31) * inv31;
  }

  FragF Oacc[4];
  #pragma unroll
  for (int t = 0; t < 4; t++)
    #pragma unroll
    for (int e = 0; e < 8; e++) Oacc[t].f[e] = 0.0f;
  float mrun[8], lrun[8];
  #pragma unroll
  for (int v = 0; v < 8; v++) { mrun[v] = -1e30f; lrun[v] = 0.0f; }

  for (int kb0 = 0; kb0 < N_TOK; kb0 += 32) {
    __syncthreads();                                   // WAR on V tile across iters
#ifdef USE_TDM
    // Tensor Data Mover: async DMA of the 32x64 bf16 V tile into LDS.
    // Wave 0 issues the descriptor, waits TENSORcnt==0, barrier releases the rest.
    if (w == 0) {
      unsigned long long ga = (unsigned long long)(uintptr_t)(Vh + (long)kb0 * D_);
      unsigned int lds_off  = (unsigned int)(uintptr_t)(&Vraw[0][0]);   // low 32 = LDS offset
      u32x4 g0;
      g0[0] = 1u;                                        // count=1, user descriptor
      g0[1] = lds_off;                                   // lds_addr (bytes)
      g0[2] = (unsigned int)(ga & 0xFFFFFFFFu);          // global_addr[31:0]
      g0[3] = (unsigned int)((ga >> 32) & 0x01FFFFFFu)   // global_addr[56:32]
              | (2u << 30);                              // type=2 ("image")
      i32x8 g1;
      g1[0] = (int)(1u << 16);                 // workgroup_mask=0, data_size=1 (2 bytes)
      g1[1] = (int)((unsigned)D_ << 16);       // tensor_dim0 = 64  (bits 79:48, low half)
      g1[2] = (int)((unsigned)N_TOK << 16);    // tensor_dim1 = 1024 (bits 111:80, low half)
      g1[3] = (int)((unsigned)D_ << 16);       // tile_dim0 = 64 (bits 127:112)
      g1[4] = 32;                              // tile_dim1 = 32 (bits 143:128)
      g1[5] = D_;                              // tensor_dim0_stride = 64 (bits 207:160)
      g1[6] = 0;
      g1[7] = 0;
      i32x4 gz = {0, 0, 0, 0};
#if __clang_major__ >= 23
      i32x8 gz8 = {0, 0, 0, 0, 0, 0, 0, 0};
      __builtin_amdgcn_tensor_load_to_lds(g0, g1, gz, gz, gz8, 0);
#else
      __builtin_amdgcn_tensor_load_to_lds(g0, g1, gz, gz, 0);
#endif
      __builtin_amdgcn_s_wait_tensorcnt(0);
    }
#else
    #pragma unroll
    for (int t = 0; t < 8; t++) {
      int lin = tid + 256 * t;                         // 0..2047
      int key = lin >> 6;
      int d   = lin & 63;
      Vt[d][key] = Vh[(long)(kb0 + key) * D_ + d];
    }
#endif
    __syncthreads();

    // S = Q * K^T   (K's row-major (N,D) layout == B-frag layout for K^T)
    FragF s0, s1;
    #pragma unroll
    for (int e = 0; e < 8; e++) { s0.f[e] = 0.0f; s1.f[e] = 0.0f; }
    #pragma unroll
    for (int ks = 0; ks < 2; ks++) {
      FragBF kf0 = load_frag(Kh, D_, kb0 + lm,      ks * 32, half);
      FragBF kf1 = load_frag(Kh, D_, kb0 + 16 + lm, ks * 32, half);
      s0.v = wmma_bf16(qa[ks], kf0, s0.v);
      s1.v = wmma_bf16(qa[ks], kf1, s1.v);
    }

    // temperature scale + analytic locality bias  loc = -d2/2
    float ym0 = (float)((kb0 + lm) >> 5) * inv31;
    float xm0 = (float)((kb0 + lm) & 31) * inv31;
    float ym1 = (float)((kb0 + 16 + lm) >> 5) * inv31;
    float xm1 = (float)((kb0 + 16 + lm) & 31) * inv31;
    #pragma unroll
    for (int v = 0; v < 8; v++) {
      float dy0 = yi[v] - ym0, dx0 = xi[v] - xm0;
      float dy1 = yi[v] - ym1, dx1 = xi[v] - xm1;
      s0.f[v] = s0.f[v] * scale - lw * 0.5f * (dy0 * dy0 + dx0 * dx0);
      s1.f[v] = s1.f[v] * scale - lw * 0.5f * (dy1 * dy1 + dx1 * dx1);
    }

    // online softmax (row spans 16 lanes within a half-wave)
    float alpha[8];
    #pragma unroll
    for (int v = 0; v < 8; v++) {
      float rm = fmaxf(s0.f[v], s1.f[v]);
      #pragma unroll
      for (int m = 1; m < 16; m <<= 1) rm = fmaxf(rm, __shfl_xor(rm, m, 32));
      float mnew = fmaxf(mrun[v], rm);
      alpha[v] = __expf(mrun[v] - mnew);
      mrun[v] = mnew;
      float p0 = __expf(s0.f[v] - mnew);
      float p1 = __expf(s1.f[v] - mnew);
      s0.f[v] = p0; s1.f[v] = p1;
      float rs = p0 + p1;
      #pragma unroll
      for (int m = 1; m < 16; m <<= 1) rs += __shfl_xor(rs, m, 32);
      lrun[v] = lrun[v] * alpha[v] + rs;
    }
    #pragma unroll
    for (int t = 0; t < 4; t++)
      #pragma unroll
      for (int v = 0; v < 8; v++) Oacc[t].f[v] *= alpha[v];

    // P: C-layout -> row-major LDS -> A-frag (per-wave region, DS in-order)
    #pragma unroll
    for (int v = 0; v < 8; v++) {
      Pbuf[w][v + 8 * half][lm]      = f2bf(s0.f[v]);
      Pbuf[w][v + 8 * half][lm + 16] = f2bf(s1.f[v]);
    }
    // A-frag reload: groups are contiguous ushorts -> two 16B LDS loads
    FragBF pa = load_frag_lds_row(&Pbuf[w][lm][0], half);

    // O += P @ V   (B-frags gathered from LDS V tile)
    #pragma unroll
    for (int t = 0; t < 4; t++) {
      FragBF vb;
#ifdef USE_TDM
      #pragma unroll
      for (int e = 0; e < 16; e++) {
        int kk = (e >> 3) * 16 + half * 8 + (e & 7);
        vb.s[e] = Vraw[kk][t * 16 + lm];
      }
#else
      vb = load_frag_lds_row(&Vt[t * 16 + lm][0], half);
#endif
      Oacc[t].v = wmma_bf16(pa, vb, Oacc[t].v);
    }
  }

  // write attention output directly in (B, N, C) bf16 for the proj GEMM
  #pragma unroll
  for (int t = 0; t < 4; t++) {
    #pragma unroll
    for (int v = 0; v < 8; v++) {
      int n = qbase + v + 8 * half;
      int c = h * D_ + t * 16 + lm;
      out[((long)b * N_TOK + n) * C_ + c] = f2bf(Oacc[t].f[v] / lrun[v]);
    }
  }
}

// ---------------- host orchestration -------------------------------------------
extern "C" void kernel_launch(void* const* d_in, const int* in_sizes, int n_in,
                              void* d_out, int out_size, void* d_ws, size_t ws_size,
                              hipStream_t stream) {
  const float* x     = (const float*)d_in[0];
  const float* wqkv  = (const float*)d_in[1];
  const float* wproj = (const float*)d_in[2];
  const float* bproj = (const float*)d_in[3];
  const float* ln1g  = (const float*)d_in[4];
  const float* ln1b  = (const float*)d_in[5];
  const float* ln2g  = (const float*)d_in[6];
  const float* ln2b  = (const float*)d_in[7];
  const float* temp  = (const float*)d_in[8];
  const float* lcw   = (const float*)d_in[9];
  const float* ls1   = (const float*)d_in[10];
  const float* ls2   = (const float*)d_in[11];
  const float* w1    = (const float*)d_in[12];
  const float* b1    = (const float*)d_in[13];
  const float* w2    = (const float*)d_in[14];
  const float* b2    = (const float*)d_in[15];
  (void)in_sizes; (void)n_in; (void)out_size; (void)ws_size;

  const int M = B_ * N_TOK;   // 8192

  char* ws = (char*)d_ws;
  size_t off = 0;
  auto alloc = [&](size_t bytes) -> char* {
    char* p = ws + off;
    off += (bytes + 255) & ~(size_t)255;
    return p;
  };
  unsigned short* wqkvT  = (unsigned short*)alloc((size_t)3 * C_ * C_ * 2);
  unsigned short* wprojT = (unsigned short*)alloc((size_t)C_ * C_ * 2);
  unsigned short* w1T    = (unsigned short*)alloc((size_t)HID_ * C_ * 2);
  unsigned short* w2T    = (unsigned short*)alloc((size_t)C_ * HID_ * 2);
  unsigned short* h1     = (unsigned short*)alloc((size_t)M * C_ * 2);
  unsigned short* qb     = (unsigned short*)alloc((size_t)M * C_ * 2);
  unsigned short* kb     = (unsigned short*)alloc((size_t)M * C_ * 2);
  unsigned short* vb     = (unsigned short*)alloc((size_t)M * C_ * 2);
  unsigned short* attn_o = (unsigned short*)alloc((size_t)M * C_ * 2);
  float*          xres   = (float*)alloc((size_t)M * C_ * 4);
  unsigned short* h2     = (unsigned short*)alloc((size_t)M * C_ * 2);
  unsigned short* hid    = (unsigned short*)alloc((size_t)M * HID_ * 2);

  // weights -> bf16, transposed to [N][K]
  cvtT_kernel<<<(3 * C_ * C_) / 256, 256, 0, stream>>>(wqkv,  wqkvT,  C_,   3 * C_);
  cvtT_kernel<<<(C_ * C_) / 256,     256, 0, stream>>>(wproj, wprojT, C_,   C_);
  cvtT_kernel<<<(C_ * HID_) / 256,   256, 0, stream>>>(w1,    w1T,    C_,   HID_);
  cvtT_kernel<<<(HID_ * C_) / 256,   256, 0, stream>>>(w2,    w2T,    HID_, C_);

  // attention branch
  ln_kernel<<<M, 256, 0, stream>>>(x, ln1g, ln1b, h1);
  gemm_kernel<0><<<dim3((3 * C_) / 128, M / 64), 256, 0, stream>>>(
      h1, wqkvT, M, 3 * C_, C_, nullptr, nullptr, nullptr, nullptr, qb, kb, vb);
  attn_kernel<<<B_ * H_ * (N_TOK / 128), 256, 0, stream>>>(qb, kb, vb, temp, lcw, attn_o);
  gemm_kernel<1><<<dim3(C_ / 128, M / 64), 256, 0, stream>>>(
      attn_o, wprojT, M, C_, C_, bproj, ls1, x, xres, nullptr, nullptr, nullptr);

  // MLP branch
  ln_kernel<<<M, 256, 0, stream>>>(xres, ln2g, ln2b, h2);
  gemm_kernel<2><<<dim3(HID_ / 128, M / 64), 256, 0, stream>>>(
      h2, w1T, M, HID_, C_, b1, nullptr, nullptr, nullptr, hid, nullptr, nullptr);
  gemm_kernel<1><<<dim3(C_ / 128, M / 64), 256, 0, stream>>>(
      hid, w2T, M, C_, HID_, b2, ls2, xres, (float*)d_out, nullptr, nullptr, nullptr);
}
